// DifferentiableDecoder_75634374082605
// MI455X (gfx1250) — matrix operand order repaired and checked by reading, not compiled, not run on hardware
//
#include <hip/hip_runtime.h>

// ---------- types ----------
typedef _Float16 h4  __attribute__((ext_vector_type(4)));
typedef _Float16 h8  __attribute__((ext_vector_type(8)));
typedef _Float16 v16h __attribute__((ext_vector_type(16)));
typedef float    v8f  __attribute__((ext_vector_type(8)));

union F16Frag { v16h v; h8 h[2]; };

// ---------- problem dims (match reference constants) ----------
static constexpr int V_ = 1024;   // vocab
static constexpr int E_ = 512;    // embedding
static constexpr int N_ = 4096;   // batch
static constexpr int M_STEPS = 32;

// ---------- GEMM tiling ----------
static constexpr int TM = 128;     // block rows
static constexpr int TN = 128;     // block cols
static constexpr int TK = 64;      // staged K-chunk = 2 WMMA K-steps
static constexpr int LDT = TK + 8; // LDS row stride in halves (144B, conflict-spread)

// C[M,N] = A[M,K] @ B[N,K]^T (+bias).  A,B fp16 row-major; C f32 (EPI=0,+bias)
// or f16 (EPI=1, no bias).  Dims are multiples of tile sizes here
// (M=4096, N in {512,1024,1536}, K in {512,1024}).
//
// Pipeline: double-buffered LDS tiles filled with GLOBAL_LOAD_ASYNC_TO_LDS_B128
// (ASYNCcnt) so chunk k+1 streams in while WMMAs consume chunk k.
template <int EPI>
__global__ __launch_bounds__(256)
void gemm_wmma(const _Float16* __restrict__ A, const _Float16* __restrict__ B,
               const float* __restrict__ bias, void* __restrict__ Cv,
               int M, int N, int K)
{
    __shared__ _Float16 As[2][TM * LDT];
    __shared__ _Float16 Bs[2][TN * LDT];

    const int tid     = threadIdx.x;
    const int lane    = tid & 31;
    const int wave    = tid >> 5;
    const int wm      = wave >> 2;     // 0..1 -> 64-row slab
    const int wn      = wave & 3;      // 0..3 -> 32-col slab
    const int laneRow = lane & 15;
    const int laneHi  = lane >> 4;     // 0/1

    const int rowBase = blockIdx.y * TM;
    const int colBase = blockIdx.x * TN;

    // Issue one 128x64-half tile of A and of B into LDS buffer `buf`,
    // 16B per lane per instruction, 4+4 async instructions per thread.
    auto issue = [&](int k0, int buf) {
#pragma unroll
        for (int i = 0; i < 4; ++i) {
            const int vid = tid + 256 * i;   // 1024 vectors of 8 halves per tile
            const int r   = vid >> 3;        // 0..127
            const int seg = (vid & 7) * 8;   // 0..56
            const _Float16* ga = A + (size_t)(rowBase + r) * K + k0 + seg;
            const _Float16* gb = B + (size_t)(colBase + r) * K + k0 + seg;
            const unsigned la = (unsigned)(size_t)&As[buf][r * LDT + seg];
            const unsigned lb = (unsigned)(size_t)&Bs[buf][r * LDT + seg];
            asm volatile("global_load_async_to_lds_b128 %0, %1, off"
                         :: "v"(la), "v"(ga) : "memory");
            asm volatile("global_load_async_to_lds_b128 %0, %1, off"
                         :: "v"(lb), "v"(gb) : "memory");
        }
    };

    v8f acc[4][2];
#pragma unroll
    for (int mt = 0; mt < 4; ++mt)
#pragma unroll
        for (int nt = 0; nt < 2; ++nt)
            acc[mt][nt] = (v8f){0.f,0.f,0.f,0.f,0.f,0.f,0.f,0.f};

    const int nchunks = K / TK;

    issue(0, 0);
    asm volatile("s_wait_asynccnt 0" ::: "memory");
    __syncthreads();

    for (int kc = 0; kc < nchunks; ++kc) {
        const int cur = kc & 1;
        if (kc + 1 < nchunks) issue((kc + 1) * TK, cur ^ 1);

        const _Float16* AT = As[cur];
        const _Float16* BT = Bs[cur];
#pragma unroll
        for (int s = 0; s < 2; ++s) {        // two WMMA K-steps per staged chunk
            const int ko = s * 32;
            // B fragments (32x16 f16): lane-hi picks K half, 16 contiguous halves
            F16Frag bf[2];
#pragma unroll
            for (int nt = 0; nt < 2; ++nt) {
                const int c = wn * 32 + nt * 16 + laneRow;
                bf[nt].h[0] = *(const h8*)&BT[c * LDT + ko + laneHi * 16];
                bf[nt].h[1] = *(const h8*)&BT[c * LDT + ko + laneHi * 16 + 8];
            }
            // A fragments (16x32 f16): lane-hi picks K half, 16 contiguous halves
#pragma unroll
            for (int mt = 0; mt < 4; ++mt) {
                const int r = wm * 64 + mt * 16 + laneRow;
                F16Frag af;
                af.h[0] = *(const h8*)&AT[r * LDT + ko + laneHi * 16];
                af.h[1] = *(const h8*)&AT[r * LDT + ko + laneHi * 16 + 8];
#pragma unroll
                for (int nt = 0; nt < 2; ++nt) {
                    acc[mt][nt] = __builtin_amdgcn_wmma_f32_16x16x32_f16(
                        false, af.v, false, bf[nt].v,
                        (short)0, acc[mt][nt], false, false);
                }
            }
        }
        if (kc + 1 < nchunks) {
            // own chunk-(kc+1) loads done, then all waves' loads visible
            asm volatile("s_wait_asynccnt 0" ::: "memory");
            __syncthreads();
        }
    }

    // ---- epilogue: C/D layout -> lane holds col=laneRow, rows laneHi*8 + r
#pragma unroll
    for (int mt = 0; mt < 4; ++mt) {
#pragma unroll
        for (int nt = 0; nt < 2; ++nt) {
            const int col = colBase + wn * 32 + nt * 16 + laneRow;
            float bv = 0.f;
            if constexpr (EPI == 0) bv = bias[col];
#pragma unroll
            for (int r2 = 0; r2 < 8; ++r2) {
                const int row = rowBase + wm * 64 + mt * 16 + laneHi * 8 + r2;
                const float val = acc[mt][nt][r2] + bv;
                if constexpr (EPI == 0)
                    ((float*)Cv)[(size_t)row * N + col] = val;
                else
                    ((_Float16*)Cv)[(size_t)row * N + col] = (_Float16)val;
            }
        }
    }
}

// ---------- f32 -> f16 weight conversion ----------
__global__ __launch_bounds__(256)
void f32_to_f16(const float* __restrict__ s, _Float16* __restrict__ d, int n)
{
    const int i = blockIdx.x * 256 + threadIdx.x;
    if (i < n) d[i] = (_Float16)s[i];
}

// ---------- init: h = x (f32 + f16 copy), probs_f16 = 0 ----------
__global__ __launch_bounds__(256)
void init_state(const float* __restrict__ x, float* __restrict__ h,
                _Float16* __restrict__ h_h, _Float16* __restrict__ probs_h)
{
    const int i = blockIdx.x * 256 + threadIdx.x;
    if (i < N_ * E_) { const float v = x[i]; h[i] = v; h_h[i] = (_Float16)v; }
    if (i < N_ * V_) probs_h[i] = (_Float16)0.f;
}

// ---------- GRU gate fusion (torch GRUCell semantics) ----------
__global__ __launch_bounds__(256)
void gru_gate(const float* __restrict__ gx, const float* __restrict__ gh,
              float* __restrict__ h, _Float16* __restrict__ h_h)
{
    const int i = blockIdx.x * 256 + threadIdx.x;
    if (i >= N_ * E_) return;
    const int n = i >> 9;          // / E_
    const int e = i & (E_ - 1);
    const float* px = gx + (size_t)n * 3 * E_;
    const float* ph = gh + (size_t)n * 3 * E_;
    const float xr = px[e],          hr = ph[e];
    const float xz = px[E_ + e],     hz = ph[E_ + e];
    const float xn = px[2 * E_ + e], hn = ph[2 * E_ + e];
    const float r  = 1.f / (1.f + __expf(-(xr + hr)));
    const float z  = 1.f / (1.f + __expf(-(xz + hz)));
    const float nn = tanhf(xn + r * hn);
    const float hv = (1.f - z) * nn + z * h[i];
    h[i]   = hv;
    h_h[i] = (_Float16)hv;
}

// ---------- row softmax over V_=1024; in-place f32 + f16 side copy ----------
__global__ __launch_bounds__(256)
void softmax_row(float* __restrict__ logits, _Float16* __restrict__ probs_h)
{
    __shared__ float red[256];
    const int row = blockIdx.x;
    const int t   = threadIdx.x;
    float* p      = logits  + (size_t)row * V_;
    _Float16* q   = probs_h + (size_t)row * V_;

    float4 x = *(const float4*)(p + t * 4);
    float m = fmaxf(fmaxf(x.x, x.y), fmaxf(x.z, x.w));
    red[t] = m; __syncthreads();
#pragma unroll
    for (int s = 128; s > 0; s >>= 1) {
        if (t < s) red[t] = fmaxf(red[t], red[t + s]);
        __syncthreads();
    }
    m = red[0]; __syncthreads();

    x.x = __expf(x.x - m); x.y = __expf(x.y - m);
    x.z = __expf(x.z - m); x.w = __expf(x.w - m);
    red[t] = x.x + x.y + x.z + x.w; __syncthreads();
#pragma unroll
    for (int s = 128; s > 0; s >>= 1) {
        if (t < s) red[t] += red[t + s];
        __syncthreads();
    }
    const float inv = 1.0f / red[0];
    x.x *= inv; x.y *= inv; x.z *= inv; x.w *= inv;
    *(float4*)(p + t * 4) = x;
    *(h4*)(q + t * 4) = (h4){(_Float16)x.x, (_Float16)x.y, (_Float16)x.z, (_Float16)x.w};
}

// ---------- host launch ----------
extern "C" void kernel_launch(void* const* d_in, const int* in_sizes, int n_in,
                              void* d_out, int out_size, void* d_ws, size_t ws_size,
                              hipStream_t stream)
{
    // setup_inputs order: x, max_len, W_d2e, W_ih, W_hh, b_ih, b_hh, W_e2d, b_e2d
    const float* x    = (const float*)d_in[0];
    const float* Wd2e = (const float*)d_in[2];
    const float* Wih  = (const float*)d_in[3];
    const float* Whh  = (const float*)d_in[4];
    const float* bih  = (const float*)d_in[5];
    const float* bhh  = (const float*)d_in[6];
    const float* We2d = (const float*)d_in[7];
    const float* be2d = (const float*)d_in[8];
    float* out        = (float*)d_out;

    char* w = (char*)d_ws;
    auto take = [&](size_t bytes) -> void* {
        void* p = (void*)w;
        w += (bytes + 255) & ~(size_t)255;
        return p;
    };
    _Float16* Wd2e_h  = (_Float16*)take((size_t)E_ * V_ * 2);
    _Float16* Wih_h   = (_Float16*)take((size_t)3 * E_ * E_ * 2);
    _Float16* Whh_h   = (_Float16*)take((size_t)3 * E_ * E_ * 2);
    _Float16* We2d_h  = (_Float16*)take((size_t)V_ * E_ * 2);
    _Float16* emb_h   = (_Float16*)take((size_t)N_ * E_ * 2);
    _Float16* h_h     = (_Float16*)take((size_t)N_ * E_ * 2);
    _Float16* probs_h = (_Float16*)take((size_t)N_ * V_ * 2);
    float*    h_f     = (float*)take((size_t)N_ * E_ * 4);
    float*    gx      = (float*)take((size_t)N_ * 3 * E_ * 4);
    float*    gh      = (float*)take((size_t)N_ * 3 * E_ * 4);

    const dim3 blk(256);

    // one-time (per launch) weight conversion to fp16
    f32_to_f16<<<(E_ * V_ + 255) / 256, blk, 0, stream>>>(Wd2e, Wd2e_h, E_ * V_);
    f32_to_f16<<<(3 * E_ * E_ + 255) / 256, blk, 0, stream>>>(Wih, Wih_h, 3 * E_ * E_);
    f32_to_f16<<<(3 * E_ * E_ + 255) / 256, blk, 0, stream>>>(Whh, Whh_h, 3 * E_ * E_);
    f32_to_f16<<<(V_ * E_ + 255) / 256, blk, 0, stream>>>(We2d, We2d_h, V_ * E_);
    init_state<<<(N_ * V_ + 255) / 256, blk, 0, stream>>>(x, h_f, h_h, probs_h);

    for (int t = 0; t < M_STEPS; ++t) {
        // emb = probs @ W_d2e^T          [4096,1024]x[512,1024]^T -> f16
        gemm_wmma<1><<<dim3(E_ / TN, N_ / TM), blk, 0, stream>>>(
            probs_h, Wd2e_h, nullptr, emb_h, N_, E_, V_);
        // gx = emb @ W_ih^T + b_ih       [4096,512]x[1536,512]^T -> f32
        gemm_wmma<0><<<dim3(3 * E_ / TN, N_ / TM), blk, 0, stream>>>(
            emb_h, Wih_h, bih, gx, N_, 3 * E_, E_);
        // gh = h @ W_hh^T + b_hh
        gemm_wmma<0><<<dim3(3 * E_ / TN, N_ / TM), blk, 0, stream>>>(
            h_h, Whh_h, bhh, gh, N_, 3 * E_, E_);
        // fused GRU gates -> new h (f32 + f16)
        gru_gate<<<(N_ * E_ + 255) / 256, blk, 0, stream>>>(gx, gh, h_f, h_h);
        // logits = h @ W_e2d^T + b_e2d   -> f32 directly into output slice
        float* logits = out + (size_t)t * N_ * V_;
        gemm_wmma<0><<<dim3(V_ / TN, N_ / TM), blk, 0, stream>>>(
            h_h, We2d_h, be2d, logits, N_, V_, E_);
        // softmax in-place + fp16 probs for next step
        softmax_row<<<N_, blk, 0, stream>>>(logits, probs_h);
    }

    (void)in_sizes; (void)n_in; (void)out_size; (void)ws_size;
}
